// TopKRouter_154618823174
// MI455X (gfx1250) — matrix operand order repaired and checked by reading, not compiled, not run on hardware
//
#include <hip/hip_runtime.h>
#include <hip/hip_bf16.h>

#define N_TOKENS  65536
#define IN_DIM    2048
#define N_EXPERTS 64

typedef __attribute__((ext_vector_type(16))) __bf16 v16bf;
typedef __attribute__((ext_vector_type(8)))  float  v8f;
typedef __attribute__((ext_vector_type(4)))  float  f4v;

// ---------------------------------------------------------------------------
// Prologue: split W (f32, [64, 2048]) into bf16 hi + bf16 residual lo.
// Whi + Wlo (512 KB total) live in d_ws and stay resident in the 192 MB L2.
// ---------------------------------------------------------------------------
__global__ __launch_bounds__(256) void split_w_kernel(const float* __restrict__ W,
                                                      __bf16* __restrict__ whi,
                                                      __bf16* __restrict__ wlo,
                                                      int n)
{
    int i = blockIdx.x * blockDim.x + threadIdx.x;
    if (i < n) {
        float w  = W[i];
        __bf16 h = (__bf16)w;
        whi[i]   = h;
        wlo[i]   = (__bf16)(w - (float)h);
    }
}

// ---------------------------------------------------------------------------
// Fused router: logits GEMM (bf16x3 split-precision WMMA) + softmax + top-2.
// Block = 256 threads = 8 wave32s. Each wave: 16 tokens x 64 experts
// (four 16x16 C tiles), K swept in chunks of 32.
//
// Pipelining: A (x) loads for chunk k+1 are issued before the chunk-k WMMAs;
// all 8 B fragments for a chunk live in distinct VGPRs so the B loads form a
// single clause whose loadcnt waits stagger across the 4 expert tiles.
// ---------------------------------------------------------------------------
__global__ __launch_bounds__(256) void router_kernel(const float*  __restrict__ x,
                                                     const __bf16* __restrict__ Whi,
                                                     const __bf16* __restrict__ Wlo,
                                                     float* __restrict__ out_idx,
                                                     float* __restrict__ out_w,
                                                     float* __restrict__ out_p)
{
    __shared__ float lds[8 * 16 * 64];   // per-wave 16x64 logits staging (32 KB)

    const int  lane = threadIdx.x & 31;
    const int  wave = threadIdx.x >> 5;
    const int  mrow = lane & 15;         // A row / C column index
    const int  half = lane >> 4;         // lane-half selects K sub-range
    const long tok0 = ((long)blockIdx.x * 8 + wave) * 16;

    v8f acc[4] = {};                     // 4 N-tiles (64 experts)

    const float* xb = x + (size_t)(tok0 + mrow) * IN_DIM;

    // ---- software pipeline: preload A fragment for chunk 0
    // A layout (matches B): per-lane K run [k0+16h, +16)
    const float* xp = xb + half * 16;
    f4v a0 = *(const f4v*)(xp +  0);
    f4v a1 = *(const f4v*)(xp +  4);
    f4v a2 = *(const f4v*)(xp +  8);
    f4v a3 = *(const f4v*)(xp + 12);

    for (int k0 = 0; k0 < IN_DIM; k0 += 32) {
        // ---- B fragments for this chunk: 4 tiles x (hi, lo), distinct regs.
        // B lane = expert column (lane&15); 16 contiguous bf16 K values.
        v16bf bhi[4], blo[4];
        #pragma unroll
        for (int t = 0; t < 4; ++t) {
            size_t wo = (size_t)(t * 16 + mrow) * IN_DIM + (size_t)(k0 + half * 16);
            bhi[t] = *(const v16bf*)(Whi + wo);
            blo[t] = *(const v16bf*)(Wlo + wo);
        }

        // ---- convert current A chunk to bf16 hi + residual lo
        float af[16];
        #pragma unroll
        for (int j = 0; j < 4; ++j) {
            af[j]      = a0[j];
            af[4 + j]  = a1[j];
            af[8 + j]  = a2[j];
            af[12 + j] = a3[j];
        }
        v16bf ahi, alo;
        #pragma unroll
        for (int e = 0; e < 16; ++e) {
            float  f  = af[e];
            __bf16 hb = (__bf16)f;
            ahi[e] = hb;
            alo[e] = (__bf16)(f - (float)hb);
        }

        // ---- prefetch A fragment for next chunk (overlaps with WMMAs below)
        if (k0 + 32 < IN_DIM) {
            const float* xn = xb + (k0 + 32) + half * 16;
            a0 = *(const f4v*)(xn +  0);
            a1 = *(const f4v*)(xn +  4);
            a2 = *(const f4v*)(xn +  8);
            a3 = *(const f4v*)(xn + 12);
        }

        // ---- 12 WMMAs: fp32 ~= hi*hi + lo*hi + hi*lo (lo*lo ~2^-16, dropped)
        #pragma unroll
        for (int t = 0; t < 4; ++t) {
            acc[t] = __builtin_amdgcn_wmma_f32_16x16x32_bf16(
                         false, ahi, false, bhi[t], (short)0, acc[t], false, false);
            acc[t] = __builtin_amdgcn_wmma_f32_16x16x32_bf16(
                         false, alo, false, bhi[t], (short)0, acc[t], false, false);
            acc[t] = __builtin_amdgcn_wmma_f32_16x16x32_bf16(
                         false, ahi, false, blo[t], (short)0, acc[t], false, false);
        }
    }

    // ---- dump logits to LDS in [token][expert] layout
    // C layout: VGPR r <-> M = r + 8*half, lane&15 = N (expert within tile)
    float* lbase = lds + wave * (16 * 64);
    #pragma unroll
    for (int t = 0; t < 4; ++t)
        #pragma unroll
        for (int r = 0; r < 8; ++r)
            lbase[(r + 8 * half) * 64 + t * 16 + mrow] = acc[t][r];

    __syncthreads();

    // ---- softmax + top-2: lanes 0..15 each own one token (64 experts)
    if (lane < 16) {
        float*      row = lbase + lane * 64;
        const long  tok = tok0 + lane;

        // pass 1: row max
        float mx = -3.402823466e+38f;
        #pragma unroll
        for (int i = 0; i < 16; ++i) {
            f4v v = *(const f4v*)(row + 4 * i);
            mx = fmaxf(mx, fmaxf(fmaxf(v[0], v[1]), fmaxf(v[2], v[3])));
        }

        // pass 2: exp(l - max) back into LDS, running sum + top-2 scan
        float s = 0.0f;
        float b1 = -1.0f, b2 = -1.0f;
        int   i1 = 0,     i2 = 0;
        #pragma unroll
        for (int i = 0; i < 16; ++i) {
            f4v v = *(const f4v*)(row + 4 * i);
            f4v e;
            #pragma unroll
            for (int j = 0; j < 4; ++j) e[j] = __expf(v[j] - mx);
            *(f4v*)(row + 4 * i) = e;
            s += e[0] + e[1] + e[2] + e[3];
            #pragma unroll
            for (int j = 0; j < 4; ++j) {
                float ev = e[j];
                int   id = 4 * i + j;
                if (ev > b1)      { b2 = b1; i2 = i1; b1 = ev; i1 = id; }
                else if (ev > b2) { b2 = ev; i2 = id; }
            }
        }

        // pass 3: normalized probs to global
        float  inv = 1.0f / s;
        float* pp  = out_p + (size_t)tok * 64;
        #pragma unroll
        for (int i = 0; i < 16; ++i) {
            f4v e = *(const f4v*)(row + 4 * i);
            f4v p;
            #pragma unroll
            for (int j = 0; j < 4; ++j) p[j] = e[j] * inv;
            *(f4v*)(pp + 4 * i) = p;
        }

        // top-2 indices + renormalized weights (softmax denom cancels)
        float tw = b1 + b2;
        out_idx[tok * 2 + 0] = (float)i1;
        out_idx[tok * 2 + 1] = (float)i2;
        out_w[tok * 2 + 0]   = b1 / tw;
        out_w[tok * 2 + 1]   = b2 / tw;
    }
}

extern "C" void kernel_launch(void* const* d_in, const int* in_sizes, int n_in,
                              void* d_out, int out_size, void* d_ws, size_t ws_size,
                              hipStream_t stream)
{
    const float* x = (const float*)d_in[0];
    const float* W = (const float*)d_in[1];

    __bf16* whi = (__bf16*)d_ws;
    __bf16* wlo = whi + (size_t)N_EXPERTS * IN_DIM;

    float* out     = (float*)d_out;
    float* out_idx = out;                        // [N, 2] indices (as float)
    float* out_w   = out + 2 * (size_t)N_TOKENS; // [N, 2] weights
    float* out_p   = out + 4 * (size_t)N_TOKENS; // [N, 64] router probs

    const int wn = N_EXPERTS * IN_DIM;           // 131072
    split_w_kernel<<<(wn + 255) / 256, 256, 0, stream>>>(W, whi, wlo, wn);

    // 512 blocks x 8 waves x 16 tokens = 65536 tokens
    router_kernel<<<N_TOKENS / 128, 256, 0, stream>>>(x, whi, wlo,
                                                      out_idx, out_w, out_p);
}
